// AttentionBlock_Crossframe_feature_traj_53953379172610
// MI455X (gfx1250) — matrix operand order repaired and checked by reading, not compiled, not run on hardware
//
#include <hip/hip_runtime.h>
#include <hip/hip_bf16.h>
#include <stdint.h>

// ---------------------------------------------------------------------------
// CDNA5 / gfx1250 implementation.
//  - All GEMMs: v_wmma_f32_16x16x32_f16 (f16 in, f32 accumulate), wave32.
//  - Weight panels / attention K,V chunks staged into LDS with the Tensor
//    Data Mover (tensor_load_to_lds + s_wait_tensorcnt), double-buffered in
//    the attention kernel.
// ---------------------------------------------------------------------------

typedef _Float16 h16;
typedef __attribute__((ext_vector_type(16))) _Float16 v16h;
typedef __attribute__((ext_vector_type(8)))  _Float16 v8h;
typedef __attribute__((ext_vector_type(8)))  float    v8f;
typedef __attribute__((ext_vector_type(4)))  unsigned int u32x4;
typedef __attribute__((ext_vector_type(8)))  int      i32x8;
typedef __attribute__((ext_vector_type(4)))  int      i32x4;

#define WMMA(a, b, c) \
  __builtin_amdgcn_wmma_f32_16x16x32_f16(false, (a), false, (b), (short)0, (c), false, false)

// A fragment (M=16 x K=32, 16-bit): lane L(<16): row M=L, elems 0..7 = K 0..7,
// elems 8..15 = K 16..23; lane L(>=16): row M=L-16, K 8..15 and 24..31.
static __device__ __forceinline__ v16h ld_frag_a(const h16* p, bool hi) {
  p += hi ? 8 : 0;
  v8h lo = *(const v8h*)p;
  v8h hh = *(const v8h*)(p + 16);
  return __builtin_shufflevector(lo, hh, 0,1,2,3,4,5,6,7,8,9,10,11,12,13,14,15);
}

// B fragment (K=32 x N=16, 16-bit): lane L(<16): col N=L, elems 0..7 = K 0..7,
// elems 8..15 = K 16..23; lane L(>=16): col N=L-16, K 8..15 and 24..31.
static __device__ __forceinline__ v16h ld_frag_b(const h16* p, bool hi) {
  p += hi ? 8 : 0;
  v8h lo = *(const v8h*)p;
  v8h hh = *(const v8h*)(p + 16);
  return __builtin_shufflevector(lo, hh, 0,1,2,3,4,5,6,7,8,9,10,11,12,13,14,15);
}

// Workgroup-relative LDS byte offset (flat LDS-aperture address bits [31:0]).
static __device__ __forceinline__ uint32_t lds_off(const void* p) {
  return (uint32_t)(uintptr_t)p;
}

// TDM 2D tile load: global (row-major, stride0 elems of 2B) -> LDS (packed).
// D# per CDNA5 ISA 8.3/8.4: group0 = {count|flags, lds_addr, gaddr_lo,
// gaddr_hi|type=2}; group1 = {data_size=2B, dims, tile dims, stride}.
static __device__ __forceinline__ void tdm_load_2d(uint32_t lds, const void* g,
    uint32_t dim0, uint32_t dim1, uint32_t tile0, uint32_t tile1,
    uint64_t stride0) {
  uint64_t ga = (uint64_t)(uintptr_t)g;
  u32x4 g0 = {};
  g0[0] = 1u;                                            // count=1 (valid), user mode
  g0[1] = lds;                                           // lds_addr (bytes)
  g0[2] = (uint32_t)ga;                                  // global_addr[31:0]
  g0[3] = (uint32_t)((ga >> 32) & 0x01ffffffu) | (2u << 30); // addr[56:32], type=2
  i32x8 g1 = {};
  g1[0] = (int)(1u << 16);                               // data_size=1 -> 2 bytes
  g1[1] = (int)((dim0 & 0xffffu) << 16);                 // tensor_dim0[15:0]
  g1[2] = (int)((dim0 >> 16) | ((dim1 & 0xffffu) << 16));// dim0[31:16], dim1[15:0]
  g1[3] = (int)((dim1 >> 16) | ((tile0 & 0xffffu) << 16)); // dim1[31:16], tile_dim0
  g1[4] = (int)(tile1 & 0xffffu);                        // tile_dim1 (tile_dim2=0)
  g1[5] = (int)(uint32_t)stride0;                        // tensor_dim0_stride[31:0]
  g1[6] = (int)((stride0 >> 32) & 0xffffu);              // stride[47:32]
  g1[7] = 0;
  i32x4 g2 = {};
  i32x4 g3 = {};
#if __clang_major__ >= 23
  i32x8 g4 = {};
  __builtin_amdgcn_tensor_load_to_lds(g0, g1, g2, g3, g4, 0);
#else
  __builtin_amdgcn_tensor_load_to_lds(g0, g1, g2, g3, 0);
#endif
}

// ---------------------------------------------------------------------------
// Kernel 1: convert weights f32 -> f16 (qkv_w 1536x512, then proj_w 512x512)
// ---------------------------------------------------------------------------
__global__ void k_wconv(const float* __restrict__ qkvw, const float* __restrict__ projw,
                        h16* __restrict__ wh) {
  int i = blockIdx.x * 256 + threadIdx.x;
  const int QN = 1536 * 512;
  const int PN = 512 * 512;
  if (i < QN)            wh[i] = (h16)qkvw[i];
  else if (i < QN + PN)  wh[i] = (h16)projw[i - QN];
}

// ---------------------------------------------------------------------------
// Kernel 2: GroupNorm. x: [8][512][1024] f32 ch-major -> xn: [8][1024][512] f16
// ---------------------------------------------------------------------------
__global__ void k_groupnorm(const float* __restrict__ x, const float* __restrict__ w,
                            const float* __restrict__ b, h16* __restrict__ xn) {
  int n = blockIdx.y, g = blockIdx.x, tid = threadIdx.x;
  const float* xb = x + ((size_t)n * 512 + g * 16) * 1024;
  float s = 0.f, s2 = 0.f;
  for (int e = tid; e < 16384; e += 256) { float v = xb[e]; s += v; s2 += v * v; }
  __shared__ float rs[256], rs2[256];
  rs[tid] = s; rs2[tid] = s2; __syncthreads();
  for (int off = 128; off > 0; off >>= 1) {
    if (tid < off) { rs[tid] += rs[tid + off]; rs2[tid] += rs2[tid + off]; }
    __syncthreads();
  }
  float mean = rs[0] * (1.f / 16384.f);
  float var  = rs2[0] * (1.f / 16384.f) - mean * mean;
  float rstd = rsqrtf(var + 1e-5f);
  for (int e = tid; e < 16384; e += 256) {
    int c = e >> 10, p = e & 1023, ch = g * 16 + c;
    float v = (xb[e] - mean) * rstd * w[ch] + b[ch];
    xn[((size_t)n * 1024 + p) * 512 + ch] = (h16)v;
  }
}

// ---------------------------------------------------------------------------
// Kernel 3: per-position squared norms of xn (wave per position).
// ---------------------------------------------------------------------------
__global__ void k_rownorm(const h16* __restrict__ xn, float* __restrict__ nrm) {
  int lane = threadIdx.x & 31, wave = threadIdx.x >> 5;
  int p = blockIdx.x * 8 + wave;              // 0..8191
  const h16* row = xn + (size_t)p * 512 + lane * 16;
  v8h a = *(const v8h*)row;
  v8h c = *(const v8h*)(row + 8);
  float s = 0.f;
  for (int r = 0; r < 8; r++) { float u = (float)a[r], v = (float)c[r]; s += u * u + v * v; }
  for (int m = 1; m < 32; m <<= 1) s += __shfl_xor(s, m);
  if (lane == 0) nrm[p] = s;
}

// ---------------------------------------------------------------------------
// Kernel 4: fused distance GEMM + argmin (WMMA), spill-free:
// 8 accumulator tiles (128 i rows) per pass, B fragment reused across them.
// d[i,j] ~ ||curr_i||^2 - 2 <curr_i, nxt_j>; argmin over i per lane-column j.
// ---------------------------------------------------------------------------
__global__ void __launch_bounds__(32) k_dist_argmin(
    const h16* __restrict__ xn, const float* __restrict__ nrm, int* __restrict__ idx) {
  int lane = threadIdx.x & 31; bool hi = lane >= 16; int l16 = lane & 15;
  int f = blockIdx.y, Jb = blockIdx.x * 16;
  const h16* cur = xn + (size_t)f * 1024 * 512;
  const h16* nxt = xn + (size_t)(f + 1) * 1024 * 512;
  const h16* bp  = nxt + (size_t)(Jb + l16) * 512;
  const float* nr = nrm + f * 1024;
  float bv = 1e30f; int bi = 0;
  for (int ic = 0; ic < 8; ic++) {                  // 8 chunks x 128 i rows
    v8f acc[8];
    for (int t = 0; t < 8; t++) for (int r = 0; r < 8; r++) acc[t][r] = 0.f;
    for (int kk = 0; kk < 16; kk++) {
      v16h b = ld_frag_b(bp + kk * 32, hi);
      for (int t = 0; t < 8; t++) {
        int row = ic * 128 + t * 16 + l16;
        v16h a = ld_frag_a(cur + (size_t)row * 512 + kk * 32, hi);
        acc[t] = WMMA(a, b, acc[t]);
      }
    }
    for (int t = 0; t < 8; t++) {
      int ib = ic * 128 + t * 16 + (hi ? 8 : 0);
      for (int r = 0; r < 8; r++) {
        float val = nr[ib + r] - 2.f * acc[t][r];
        if (val < bv) { bv = val; bi = ib + r; }
      }
    }
  }
  float ov = __shfl_xor(bv, 16);
  int   oi = __shfl_xor(bi, 16);
  if (ov < bv || (ov == bv && oi < bi)) { bv = ov; bi = oi; }
  if (!hi) idx[f * 1024 + Jb + l16] = bi;
}

// ---------------------------------------------------------------------------
// Kernel 5: gather. xg[f][j][:] = xn[min(f+1,7)][ f<7 ? idx[f][j] : j ][:]
// ---------------------------------------------------------------------------
__global__ void k_gather(const h16* __restrict__ xn, const int* __restrict__ idx,
                         h16* __restrict__ xg) {
  int gid = blockIdx.x * 256 + threadIdx.x;      // 8*1024*64 units of 8 halves
  int c8 = gid & 63, j = (gid >> 6) & 1023, f = gid >> 16;
  int sf  = (f < 7) ? f + 1 : 7;
  int src = (f < 7) ? idx[f * 1024 + j] : j;
  const v8h* sp = (const v8h*)(xn + ((size_t)sf * 1024 + src) * 512 + c8 * 8);
  *(v8h*)(xg + ((size_t)f * 1024 + j) * 512 + c8 * 8) = *sp;
}

// ---------------------------------------------------------------------------
// Kernel 6: generic WMMA GEMM  R[o,n] = sum_c W[o,c] * X[n,c].
// Block = 4 waves, M=64 x N=64. The 64x512 f16 weight panel (64 KB) is DMA'd
// into LDS once per block with the TDM; A-fragments then come from LDS.
// mode 0: q   -> outP[n][o] = (R+bias)*scale                  (f16, ld 512)
// mode 1: k/v -> outP[(f*2048+pos+colOff)][chOff+o] scaled    (f16, ld 1024)
//                outC[f][chOff+o][pos+colOff]     unscaled    (f16, ld 2048)
// mode 2: proj-> outF[f][o][pos] = R + bias + resid           (f32)
// ---------------------------------------------------------------------------
__global__ void __launch_bounds__(128) k_gemm(
    const h16* __restrict__ W, const h16* __restrict__ X,
    const float* __restrict__ bias, int mode, float scale,
    int colOff, int chOff,
    h16* __restrict__ outP, h16* __restrict__ outC,
    float* __restrict__ outF, const float* __restrict__ resid) {
  __shared__ h16 Ws[64 * 512];                    // 64 KB weight panel
  int lane = threadIdx.x & 31, wave = threadIdx.x >> 5;
  bool hi = lane >= 16; int l16 = lane & 15;
  int n = blockIdx.y * 64 + wave * 16 + l16;      // global column (f*1024+pos)
  int f = n >> 10, pos = n & 1023;
  int mbase = blockIdx.x * 64;
  if (wave == 0) {
    tdm_load_2d(lds_off(Ws), W + (size_t)mbase * 512, 512, 64, 512, 64, 512);
    __builtin_amdgcn_s_wait_tensorcnt(0);
  }
  __syncthreads();
  v8f acc[4];
  for (int i = 0; i < 4; i++) for (int r = 0; r < 8; r++) acc[i][r] = 0.f;
  const h16* xcol = X + (size_t)n * 512;
  for (int k0 = 0; k0 < 512; k0 += 32) {
    v16h b = ld_frag_b(xcol + k0, hi);
    for (int mt = 0; mt < 4; mt++) {
      v16h a = ld_frag_a(&Ws[(mt * 16 + l16) * 512 + k0], hi);
      acc[mt] = WMMA(a, b, acc[mt]);
    }
  }
  for (int mt = 0; mt < 4; mt++) {
    int ob = mbase + mt * 16 + (hi ? 8 : 0);      // first of 8 contiguous out rows
    if (mode == 0) {
      v8h s;
      for (int r = 0; r < 8; r++) s[r] = (h16)((acc[mt][r] + bias[ob + r]) * scale);
      *(v8h*)(outP + (size_t)n * 512 + ob) = s;
    } else if (mode == 1) {
      v8h s;
      for (int r = 0; r < 8; r++) s[r] = (h16)((acc[mt][r] + bias[ob + r]) * scale);
      *(v8h*)(outP + ((size_t)(f * 2048 + pos + colOff)) * 1024 + chOff + ob) = s;
      for (int r = 0; r < 8; r++)
        outC[(size_t)f * 1024 * 2048 + (size_t)(chOff + ob + r) * 2048 + pos + colOff] =
            (h16)(acc[mt][r] + bias[ob + r]);
    } else {
      for (int r = 0; r < 8; r++) {
        size_t oi = (size_t)f * 512 * 1024 + (size_t)(ob + r) * 1024 + pos;
        outF[oi] = acc[mt][r] + bias[ob + r] + resid[oi];
      }
    }
  }
}

// ---------------------------------------------------------------------------
// Kernel 7: flash attention, TDM double-buffered.
// Block = 4 waves sharing one head (m = f*8+h); wave w handles 16 query rows.
// K chunk (32 s x 64 c) and V chunk (64 c x 32 s) are DMA'd into LDS by the
// TDM; next chunk's DMA overlaps current chunk's WMMAs (s_wait_tensorcnt(2)).
// Q: qP[f][t][h*64+c] (pre-scaled). K: kvP[f][s][h*128+c] (pre-scaled).
// V: kvC[f][h*128+64+c][s]. Out: aP[f][t][h*64+c].
// ---------------------------------------------------------------------------
__global__ void __launch_bounds__(128) k_attn(
    const h16* __restrict__ qP, const h16* __restrict__ kvP,
    const h16* __restrict__ kvC, h16* __restrict__ aP) {
  __shared__ h16 Kl[2][32 * 64];                  // [buf][s][c]
  __shared__ h16 Vl[2][64 * 32];                  // [buf][c][s]
  __shared__ h16 Pl[4][16 * 40];                  // per-wave [t][s_local], pad 40
  int lane = threadIdx.x & 31, wave = threadIdx.x >> 5;
  bool hi = lane >= 16; int l16 = lane & 15;
  int m = blockIdx.y, f = m >> 3, h = m & 7;
  int Tb = blockIdx.x * 64 + wave * 16;
  const h16* qrow = qP + ((size_t)(f * 1024 + Tb + l16)) * 512 + h * 64;
  v16h qb0 = ld_frag_b(qrow, hi);
  v16h qb1 = ld_frag_b(qrow + 32, hi);
  const h16* kbase = kvP + (size_t)f * 2048 * 1024 + h * 128;               // s stride 1024
  const h16* vbase = kvC + (size_t)f * 1024 * 2048 + (size_t)(h * 128 + 64) * 2048; // c stride 2048
  v8f aacc[4];
  for (int i = 0; i < 4; i++) for (int r = 0; r < 8; r++) aacc[i][r] = 0.f;
  float mrun = -1e30f, lrun = 0.f;
  if (wave == 0) {
    tdm_load_2d(lds_off(Kl[0]), kbase, 64, 32, 64, 32, 1024);
    tdm_load_2d(lds_off(Vl[0]), vbase, 32, 64, 32, 64, 2048);
  }
  for (int i = 0; i < 64; i++) {
    int cur = i & 1;
    if (wave == 0) {
      if (i + 1 < 64) {
        size_t s1 = (size_t)(i + 1) * 32;
        tdm_load_2d(lds_off(Kl[1 - cur]), kbase + s1 * 1024, 64, 32, 64, 32, 1024);
        tdm_load_2d(lds_off(Vl[1 - cur]), vbase + s1,        32, 64, 32, 64, 2048);
        __builtin_amdgcn_s_wait_tensorcnt(2);   // chunk i complete, i+1 in flight
      } else {
        __builtin_amdgcn_s_wait_tensorcnt(0);
      }
    }
    __syncthreads();
    // logits (transposed: M = s, N = t) for this 32-s chunk
    v8f lg[2];
    for (int mt = 0; mt < 2; mt++) {
      for (int r = 0; r < 8; r++) lg[mt][r] = 0.f;
      const h16* kr = &Kl[cur][(mt * 16 + l16) * 64];
      lg[mt] = WMMA(ld_frag_a(kr, hi),      qb0, lg[mt]);
      lg[mt] = WMMA(ld_frag_a(kr + 32, hi), qb1, lg[mt]);
    }
    float tmax = mrun;
    for (int mt = 0; mt < 2; mt++)
      for (int r = 0; r < 8; r++) tmax = fmaxf(tmax, lg[mt][r]);
    tmax = fmaxf(tmax, __shfl_xor(tmax, 16));     // share max across lane pair (same t)
    float alpha = __expf(mrun - tmax);
    lrun *= alpha;
    for (int ct = 0; ct < 4; ct++)
      for (int r = 0; r < 8; r++) aacc[ct][r] *= alpha;
    for (int mt = 0; mt < 2; mt++) {
      v8h ph;
      for (int r = 0; r < 8; r++) {
        float p = __expf(lg[mt][r] - tmax);
        lrun += p;
        ph[r] = (h16)p;
      }
      *(v8h*)(&Pl[wave][l16 * 40 + mt * 16 + (hi ? 8 : 0)]) = ph;
    }
    // per-wave LDS exchange (same-wave DS ops are in order)
    v16h pb = ld_frag_b(&Pl[wave][l16 * 40], hi); // P[K=s_local][N=t]
    for (int ct = 0; ct < 4; ct++) {
      v16h vf = ld_frag_a(&Vl[cur][(ct * 16 + l16) * 32], hi);
      aacc[ct] = WMMA(vf, pb, aacc[ct]);
    }
    mrun = tmax;
    __syncthreads();                              // done reading Kl/Vl[cur]
  }
  float ltot = lrun + __shfl_xor(lrun, 16);
  float inv = 1.f / ltot;
  for (int ct = 0; ct < 4; ct++) {
    v8h s;
    for (int r = 0; r < 8; r++) s[r] = (h16)(aacc[ct][r] * inv);
    *(v8h*)(aP + ((size_t)(f * 1024 + Tb + l16)) * 512 + h * 64 + ct * 16 + (hi ? 8 : 0)) = s;
  }
}

// ---------------------------------------------------------------------------
// Launch. Workspace layout (bytes, ~98.1 MB total):
//   XN 0..8M | XG 8..16M | QP 16..24M | KVP 24..56M | KVC 56..88M
//   AP 88..96M | WH 96..98M | NRM 98M..+32K | IDX 98M+64K..+28K
// ---------------------------------------------------------------------------
extern "C" void kernel_launch(void* const* d_in, const int* in_sizes, int n_in,
                              void* d_out, int out_size, void* d_ws, size_t ws_size,
                              hipStream_t stream) {
  (void)in_sizes; (void)n_in; (void)out_size; (void)ws_size;
  const float* x      = (const float*)d_in[0];
  const float* norm_w = (const float*)d_in[2];
  const float* norm_b = (const float*)d_in[3];
  const float* qkv_w  = (const float*)d_in[4];
  const float* qkv_b  = (const float*)d_in[5];
  const float* proj_w = (const float*)d_in[6];
  const float* proj_b = (const float*)d_in[7];
  float* out = (float*)d_out;
  char* ws = (char*)d_ws;
  const size_t MB = 1024u * 1024u;
  h16*   XN  = (h16*)(ws + 0 * MB);
  h16*   XG  = (h16*)(ws + 8 * MB);
  h16*   QP  = (h16*)(ws + 16 * MB);
  h16*   KVP = (h16*)(ws + 24 * MB);
  h16*   KVC = (h16*)(ws + 56 * MB);
  h16*   AP  = (h16*)(ws + 88 * MB);
  h16*   WH  = (h16*)(ws + 96 * MB);
  float* NRM = (float*)(ws + 98 * MB);
  int*   IDX = (int*)(ws + 98 * MB + 65536);
  const float SC = 0.35355339059327373f;   // 1/sqrt(sqrt(64))

  k_wconv<<<4096, 256, 0, stream>>>(qkv_w, proj_w, WH);
  k_groupnorm<<<dim3(32, 8), 256, 0, stream>>>(x, norm_w, norm_b, XN);
  k_rownorm<<<1024, 256, 0, stream>>>(XN, NRM);
  k_dist_argmin<<<dim3(64, 7), 32, 0, stream>>>(XN, NRM, IDX);
  k_gather<<<2048, 256, 0, stream>>>(XN, IDX, XG);
  // q
  k_gemm<<<dim3(8, 128), 128, 0, stream>>>(WH,              XN, qkv_b,        0, SC, 0,    0,   QP,  nullptr, nullptr, nullptr);
  // k (kv channels 0..511): first-half cols from xn, second-half from gather
  k_gemm<<<dim3(8, 128), 128, 0, stream>>>(WH + 512 * 512,  XN, qkv_b + 512,  1, SC, 0,    0,   KVP, KVC,     nullptr, nullptr);
  k_gemm<<<dim3(8, 128), 128, 0, stream>>>(WH + 512 * 512,  XG, qkv_b + 512,  1, SC, 1024, 0,   KVP, KVC,     nullptr, nullptr);
  // v (kv channels 512..1023)
  k_gemm<<<dim3(8, 128), 128, 0, stream>>>(WH + 1024 * 512, XN, qkv_b + 1024, 1, SC, 0,    512, KVP, KVC,     nullptr, nullptr);
  k_gemm<<<dim3(8, 128), 128, 0, stream>>>(WH + 1024 * 512, XG, qkv_b + 1024, 1, SC, 1024, 512, KVP, KVC,     nullptr, nullptr);
  k_attn<<<dim3(16, 64), 128, 0, stream>>>(QP, KVP, KVC, AP);
  // proj + bias + residual -> d_out (f32)
  k_gemm<<<dim3(8, 128), 128, 0, stream>>>(WH + 1536 * 512, AP, proj_b,       2, 1.f, 0,   0,   nullptr, nullptr, out, x);
}
